// SetAttentionLayer_38903813767401
// MI455X (gfx1250) — matrix operand order repaired and checked by reading, not compiled, not run on hardware
//
#include <hip/hip_runtime.h>
#include <hip/hip_bf16.h>

typedef __attribute__((ext_vector_type(16))) _Float16 v16h;
typedef __attribute__((ext_vector_type(8)))  _Float16 v8h;
typedef __attribute__((ext_vector_type(8)))  float    v8f;
typedef __attribute__((ext_vector_type(4)))  float    v4f;

#define HPAD 136   // LDS row stride in halves for 128-wide tiles (16B-aligned chunks)

__device__ __forceinline__ int imin(int a, int b) { return a < b ? a : b; }

// order-preserving float<->uint encoding for atomicMax-based segment max
__device__ __forceinline__ unsigned fenc(float f) {
    unsigned u = __float_as_uint(f);
    return (u & 0x80000000u) ? ~u : (u | 0x80000000u);
}
__device__ __forceinline__ float fdec(unsigned u) {
    return (u & 0x80000000u) ? __uint_as_float(u & 0x7FFFFFFFu) : __uint_as_float(~u);
}

// ---------------- workspace zeroing ----------------
__global__ void zero_kernel(float* __restrict__ p, size_t n) {
    for (size_t i = blockIdx.x * (size_t)blockDim.x + threadIdx.x; i < n;
         i += (size_t)gridDim.x * blockDim.x)
        p[i] = 0.0f;
}

// ---------------- weight -> WMMA B-fragment swizzle ----------------
// dst layout: frag (kc*8+nt): 32 lanes x 16 halves; lane element i holds
// W[(kc*32 + (i<8 ? 8*(lane>>4)+i : 16+8*(lane>>4)+i-8))*Nout + nt*16 + (lane&15)]  (B is K x Nout row-major)
__global__ void prep_frags_kernel(const float* __restrict__ W, _Float16* __restrict__ dst,
                                  int kchunks, int Nout) {
    int total = kchunks * 8 * 512;
    for (int e = blockIdx.x * blockDim.x + threadIdx.x; e < total;
         e += gridDim.x * blockDim.x) {
        int i    = e & 15;
        int lane = (e >> 4) & 31;
        int nt   = (e >> 9) & 7;
        int kc   = e >> 12;
        int half = lane >> 4;
        int n    = nt * 16 + (lane & 15);
        int K    = kc * 32 + (i < 8 ? 8 * half + i : 16 + 8 * half + (i - 8));
        dst[e] = (_Float16)W[(size_t)K * Nout + n];
    }
}

// ---------------- collapse attention GEMM: w_eff[160][4] ----------------
__global__ void weff_kernel(const float* __restrict__ Wk, const float* __restrict__ Wq,
                            float* __restrict__ weff) {
    int t = blockIdx.x * blockDim.x + threadIdx.x;
    if (t >= 640) return;
    int i = t >> 2, h = t & 3;
    float s = 0.0f;
    for (int d = 0; d < 64; ++d)
        s += Wk[(size_t)i * 256 + h * 64 + d] * Wq[h * 64 + d];
    weff[t] = s * 0.125f;   // fold 1/sqrt(64)
}

// ---------------- psi MLP (3x WMMA GEMM) + fused segment-sum pooling ----------------
__global__ __launch_bounds__(128)
void mlp_pool_kernel(const float* __restrict__ inputs, const int* __restrict__ seg,
                     const float* __restrict__ b1, const float* __restrict__ b2,
                     const float* __restrict__ b3,
                     const _Float16* __restrict__ w1f, const _Float16* __restrict__ w2f,
                     const _Float16* __restrict__ w3f,
                     float* __restrict__ seg_sum, float* __restrict__ counts, int N) {
    __shared__ _Float16 htile[4][16 * HPAD];   // per-wave f16 activation staging
    __shared__ float    hbuf[64 * 129];        // block-wide f32 h for pooling
    __shared__ int      sseg[64];

    const int lane = threadIdx.x & 31;
    const int wid  = threadIdx.x >> 5;
    const int half = lane >> 4;
    const int m    = lane & 15;
    const int blockRow0 = blockIdx.x * 64;
    const int base = blockRow0 + wid * 16;
    const int row  = imin(base + m, N - 1);

    // ---- A1 fragment straight from global (K=32 == one fragment) ----
    v16h a1;
    {
        const float* p = inputs + (size_t)row * 32 + 8 * half;
        v4f x0 = *(const v4f*)(p);
        v4f x1 = *(const v4f*)(p + 4);
        v4f x2 = *(const v4f*)(p + 16);
        v4f x3 = *(const v4f*)(p + 20);
#pragma unroll
        for (int j = 0; j < 4; ++j) {
            a1[j]      = (_Float16)x0[j];
            a1[4 + j]  = (_Float16)x1[j];
            a1[8 + j]  = (_Float16)x2[j];
            a1[12 + j] = (_Float16)x3[j];
        }
    }

    v8f acc[8];
    // ---- GEMM1: [16,32] x [32,128] ----
#pragma unroll
    for (int nt = 0; nt < 8; ++nt) {
        const v16h b = *(const v16h*)(w1f + (((size_t)nt * 32 + lane) << 4));
        v8f c = {};
        acc[nt] = __builtin_amdgcn_wmma_f32_16x16x32_f16(false, a1, false, b,
                                                         (short)0, c, false, false);
    }
    _Float16* ht = htile[wid];
#pragma unroll
    for (int nt = 0; nt < 8; ++nt) {
        const float bias = b1[nt * 16 + m];
#pragma unroll
        for (int r = 0; r < 8; ++r) {
            float v = acc[nt][r] + bias;
            ht[(r + 8 * half) * HPAD + nt * 16 + m] = (_Float16)(v > 0.0f ? v : 0.0f);
        }
    }
    __syncthreads();

    // ---- GEMM2: [16,128] x [128,128] ----
#pragma unroll
    for (int nt = 0; nt < 8; ++nt) { v8f z = {}; acc[nt] = z; }
#pragma unroll
    for (int kc = 0; kc < 4; ++kc) {
        const _Float16* pr = ht + m * HPAD + kc * 32 + 8 * half;
        v8h lo = *(const v8h*)pr;
        v8h hi = *(const v8h*)(pr + 16);
        v16h a2 = __builtin_shufflevector(lo, hi, 0,1,2,3,4,5,6,7,8,9,10,11,12,13,14,15);
#pragma unroll
        for (int nt = 0; nt < 8; ++nt) {
            const v16h b = *(const v16h*)(w2f + ((((size_t)kc * 8 + nt) * 32 + lane) << 4));
            acc[nt] = __builtin_amdgcn_wmma_f32_16x16x32_f16(false, a2, false, b,
                                                             (short)0, acc[nt], false, false);
        }
    }
    __syncthreads();
#pragma unroll
    for (int nt = 0; nt < 8; ++nt) {
        const float bias = b2[nt * 16 + m];
#pragma unroll
        for (int r = 0; r < 8; ++r) {
            float v = acc[nt][r] + bias;
            ht[(r + 8 * half) * HPAD + nt * 16 + m] = (_Float16)(v > 0.0f ? v : 0.0f);
        }
    }
    __syncthreads();

    // ---- GEMM3: [16,128] x [128,128] (linear) ----
#pragma unroll
    for (int nt = 0; nt < 8; ++nt) { v8f z = {}; acc[nt] = z; }
#pragma unroll
    for (int kc = 0; kc < 4; ++kc) {
        const _Float16* pr = ht + m * HPAD + kc * 32 + 8 * half;
        v8h lo = *(const v8h*)pr;
        v8h hi = *(const v8h*)(pr + 16);
        v16h a3 = __builtin_shufflevector(lo, hi, 0,1,2,3,4,5,6,7,8,9,10,11,12,13,14,15);
#pragma unroll
        for (int nt = 0; nt < 8; ++nt) {
            const v16h b = *(const v16h*)(w3f + ((((size_t)kc * 8 + nt) * 32 + lane) << 4));
            acc[nt] = __builtin_amdgcn_wmma_f32_16x16x32_f16(false, a3, false, b,
                                                             (short)0, acc[nt], false, false);
        }
    }
    // ---- stage f32 h into block-wide LDS for segmented pooling ----
#pragma unroll
    for (int nt = 0; nt < 8; ++nt) {
        const float bias = b3[nt * 16 + m];
#pragma unroll
        for (int r = 0; r < 8; ++r)
            hbuf[(wid * 16 + r + 8 * half) * 129 + nt * 16 + m] = acc[nt][r] + bias;
    }
    if (threadIdx.x < 64)
        sseg[threadIdx.x] = seg[imin(blockRow0 + (int)threadIdx.x, N - 1)];
    __syncthreads();

    // ---- segmented reduction: one thread per column, runs over sorted ids ----
    {
        const int c = threadIdx.x;   // 0..127
        const int rows = imin(64, N - blockRow0);
        int   cur = sseg[0];
        float s   = 0.0f;
        int   len = 0;
        for (int r = 0; r < rows; ++r) {
            int sg = sseg[r];
            if (sg != cur) {
                atomicAdd(&seg_sum[(size_t)cur * 128 + c], s);
                if (c == 0) atomicAdd(&counts[cur], (float)len);
                cur = sg; s = 0.0f; len = 0;
            }
            s += hbuf[r * 129 + c];
            ++len;
        }
        atomicAdd(&seg_sum[(size_t)cur * 128 + c], s);
        if (c == 0) atomicAdd(&counts[cur], (float)len);
    }
}

// ---------------- rho: agg = (seg_sum/count) @ Wr + br  (WMMA) ----------------
__global__ __launch_bounds__(128)
void rho_kernel(const float* __restrict__ seg_sum, const float* __restrict__ counts,
                const float* __restrict__ br, const _Float16* __restrict__ wrf,
                float* __restrict__ agg2, int B) {
    const int tile = blockIdx.x * 4 + (threadIdx.x >> 5);   // wave-uniform
    const int lane = threadIdx.x & 31;
    const int half = lane >> 4;
    const int m    = lane & 15;
    const int base = tile * 16;
    if (base >= B) return;
    const int rowm = imin(base + m, B - 1);
    const float inv = 1.0f / fmaxf(counts[rowm], 1.0f);
    const float* srow = seg_sum + (size_t)rowm * 128;

    v8f acc[8];
#pragma unroll
    for (int nt = 0; nt < 8; ++nt) { v8f z = {}; acc[nt] = z; }
#pragma unroll
    for (int kc = 0; kc < 4; ++kc) {
        const float* p = srow + kc * 32 + 8 * half;
        v16h a;
#pragma unroll
        for (int j = 0; j < 8; ++j) {
            a[j]     = (_Float16)(p[j] * inv);
            a[8 + j] = (_Float16)(p[16 + j] * inv);
        }
#pragma unroll
        for (int nt = 0; nt < 8; ++nt) {
            const v16h b = *(const v16h*)(wrf + ((((size_t)kc * 8 + nt) * 32 + lane) << 4));
            acc[nt] = __builtin_amdgcn_wmma_f32_16x16x32_f16(false, a, false, b,
                                                             (short)0, acc[nt], false, false);
        }
    }
#pragma unroll
    for (int nt = 0; nt < 8; ++nt) {
        const float bias = br[nt * 16 + m];
#pragma unroll
        for (int r = 0; r < 8; ++r) {
            int orow = base + r + 8 * half;
            if (orow < B) agg2[(size_t)orow * 128 + nt * 16 + m] = acc[nt][r] + bias;
        }
    }
}

// ---------------- preattn = [x, agg[seg]] @ w_eff  + segment atomicMax ----------------
__global__ __launch_bounds__(256)
void preattn_kernel(const float* __restrict__ inputs, const int* __restrict__ seg,
                    const float* __restrict__ agg2, const float* __restrict__ weff,
                    float* __restrict__ preattn, unsigned* __restrict__ segmax, int N) {
    __shared__ float sw[640];
    for (int i = threadIdx.x; i < 640; i += blockDim.x) sw[i] = weff[i];
    __syncthreads();
    int n = blockIdx.x * blockDim.x + threadIdx.x;
    if (n >= N) return;
    const int s = seg[n];
    float a0 = 0, a1 = 0, a2 = 0, a3 = 0;
    const float* x = inputs + (size_t)n * 32;
#pragma unroll 8
    for (int i = 0; i < 32; ++i) {
        float v = x[i];
        a0 += v * sw[i * 4 + 0]; a1 += v * sw[i * 4 + 1];
        a2 += v * sw[i * 4 + 2]; a3 += v * sw[i * 4 + 3];
    }
    const float* ag = agg2 + (size_t)s * 128;
#pragma unroll 8
    for (int j = 0; j < 128; ++j) {
        float v = ag[j];
        int i = 32 + j;
        a0 += v * sw[i * 4 + 0]; a1 += v * sw[i * 4 + 1];
        a2 += v * sw[i * 4 + 2]; a3 += v * sw[i * 4 + 3];
    }
    preattn[(size_t)n * 4 + 0] = a0; preattn[(size_t)n * 4 + 1] = a1;
    preattn[(size_t)n * 4 + 2] = a2; preattn[(size_t)n * 4 + 3] = a3;
    atomicMax(&segmax[(size_t)s * 4 + 0], fenc(a0));
    atomicMax(&segmax[(size_t)s * 4 + 1], fenc(a1));
    atomicMax(&segmax[(size_t)s * 4 + 2], fenc(a2));
    atomicMax(&segmax[(size_t)s * 4 + 3], fenc(a3));
}

// ---------------- ex = exp(pre - max); denom = segment_sum(ex) ----------------
__global__ __launch_bounds__(256)
void expdenom_kernel(const int* __restrict__ seg, float* __restrict__ preattn,
                     const unsigned* __restrict__ segmax, float* __restrict__ denom, int N) {
    int n = blockIdx.x * blockDim.x + threadIdx.x;
    if (n >= N) return;
    const int s = seg[n];
#pragma unroll
    for (int h = 0; h < 4; ++h) {
        float mx = fdec(segmax[(size_t)s * 4 + h]);
        float e = __expf(preattn[(size_t)n * 4 + h] - mx);
        preattn[(size_t)n * 4 + h] = e;
        atomicAdd(&denom[(size_t)s * 4 + h], e);
    }
}

// ---------------- attn = ex / denom -> out[h][n] ----------------
__global__ __launch_bounds__(256)
void norm_kernel(const int* __restrict__ seg, const float* __restrict__ ex,
                 const float* __restrict__ denom, float* __restrict__ out, int N) {
    int n = blockIdx.x * blockDim.x + threadIdx.x;
    if (n >= N) return;
    const int s = seg[n];
#pragma unroll
    for (int h = 0; h < 4; ++h)
        out[(size_t)h * N + n] = ex[(size_t)n * 4 + h] / denom[(size_t)s * 4 + h];
}

extern "C" void kernel_launch(void* const* d_in, const int* in_sizes, int n_in,
                              void* d_out, int out_size, void* d_ws, size_t ws_size,
                              hipStream_t stream) {
    const float* inputs = (const float*)d_in[0];
    const int*   seg    = (const int*)d_in[1];
    const int    N      = in_sizes[1];
    const int    B      = in_sizes[2];
    const float* W1 = (const float*)d_in[3];  const float* b1 = (const float*)d_in[4];
    const float* W2 = (const float*)d_in[5];  const float* b2 = (const float*)d_in[6];
    const float* W3 = (const float*)d_in[7];  const float* b3 = (const float*)d_in[8];
    const float* Wr = (const float*)d_in[9];  const float* br = (const float*)d_in[10];
    const float* Wk = (const float*)d_in[11]; const float* Wq = (const float*)d_in[12];
    float* out = (float*)d_out;

    // ---- workspace carve-out (256B-aligned sections) ----
    char* p = (char*)d_ws;
    auto carve = [&](size_t bytes) -> char* {
        char* r = p;
        p += (bytes + 255) & ~(size_t)255;
        return r;
    };
    float*    seg_sum = (float*)carve((size_t)B * 128 * 4);
    float*    counts  = (float*)carve((size_t)B * 4);
    float*    denom   = (float*)carve((size_t)B * 4 * 4);
    unsigned* segmax  = (unsigned*)carve((size_t)B * 4 * 4);
    char*     zero_end = p;                    // everything above must start at 0
    float*    agg2    = (float*)carve((size_t)B * 128 * 4);
    float*    preattn = (float*)carve((size_t)N * 4 * 4);
    float*    weff    = (float*)carve(640 * 4);
    _Float16* w1f     = (_Float16*)carve(4096 * 2);
    _Float16* w2f     = (_Float16*)carve(16384 * 2);
    _Float16* w3f     = (_Float16*)carve(16384 * 2);
    _Float16* wrf     = (_Float16*)carve(16384 * 2);

    const size_t zeroFloats = (size_t)(zero_end - (char*)d_ws) / 4;
    zero_kernel<<<2048, 256, 0, stream>>>((float*)d_ws, zeroFloats);

    prep_frags_kernel<<<16, 256, 0, stream>>>(W1, w1f, 1, 128);
    prep_frags_kernel<<<64, 256, 0, stream>>>(W2, w2f, 4, 128);
    prep_frags_kernel<<<64, 256, 0, stream>>>(W3, w3f, 4, 128);
    prep_frags_kernel<<<64, 256, 0, stream>>>(Wr, wrf, 4, 128);
    weff_kernel<<<3, 256, 0, stream>>>(Wk, Wq, weff);

    mlp_pool_kernel<<<(N + 63) / 64, 128, 0, stream>>>(
        inputs, seg, b1, b2, b3, w1f, w2f, w3f, seg_sum, counts, N);

    rho_kernel<<<(((B + 15) / 16) + 3) / 4, 128, 0, stream>>>(
        seg_sum, counts, br, wrf, agg2, B);

    preattn_kernel<<<(N + 255) / 256, 256, 0, stream>>>(
        inputs, seg, agg2, weff, preattn, segmax, N);
    expdenom_kernel<<<(N + 255) / 256, 256, 0, stream>>>(seg, preattn, segmax, denom, N);
    norm_kernel<<<(N + 255) / 256, 256, 0, stream>>>(seg, preattn, denom, out, N);
    (void)out_size; (void)ws_size; (void)n_in;
}
